// GeneralMultiHeadAttention_80891414053262
// MI455X (gfx1250) — matrix operand order repaired and checked by reading, not compiled, not run on hardware
//
#include <hip/hip_runtime.h>

// ---------------- problem constants ----------------
constexpr int B  = 2;
constexpr int S  = 2048;
constexpr int D  = 512;
constexpr int H  = 8;
constexpr int DK = 64;                 // D / H
constexpr float INV_TEMP = 0.125f;     // 1/sqrt(64)
constexpr float NEG = -1e9f;

typedef __attribute__((ext_vector_type(16))) _Float16 v16h;
typedef __attribute__((ext_vector_type(8)))  float    v8f;

// ---------------- WMMA wrapper ----------------
__device__ __forceinline__ v8f wmma_f16(v16h a, v16h b, v8f c) {
    // D = A(16x32 f16) * B(32x16 f16) + C(16x16 f32)
    return __builtin_amdgcn_wmma_f32_16x16x32_f16(
        /*neg_a=*/false, a, /*neg_b=*/false, b,
        /*c_mod=*/(short)0, c, /*reuse_a=*/false, /*reuse_b=*/false);
}

// ---------------- operand loaders (contiguous k = 16*half + s order, wave32) ----------------
// A-matrix 16x32 (MxK): lane row M=l&15 holds K = half*16 + g*8 + j (g=0,1; j=0..7)
__device__ __forceinline__ v16h load_a_f32(const float* base, int ld) {
    int l = threadIdx.x & 31; int m = l & 15; int hf = l >> 4;
    v16h a;
#pragma unroll
    for (int g = 0; g < 2; ++g) {
        const float* p = base + (size_t)m * ld + hf * 16 + g * 8;
#pragma unroll
        for (int j = 0; j < 8; ++j) a[g * 8 + j] = (_Float16)p[j];
    }
    return a;
}
__device__ __forceinline__ v16h load_a_f16(const _Float16* base, int ld) {
    int l = threadIdx.x & 31; int m = l & 15; int hf = l >> 4;
    v16h a;
#pragma unroll
    for (int g = 0; g < 2; ++g) {
        const _Float16* p = base + (size_t)m * ld + hf * 16 + g * 8;
#pragma unroll
        for (int j = 0; j < 8; ++j) a[g * 8 + j] = p[j];
    }
    return a;
}
// B-matrix 32x16 (KxN): lane holds column N = l&15; K = (l>>4)*16 + i, i=0..15
// source addressed as col_base[n*ld + k]  (i.e. B[k][n] = col_base[n*ld + k])
__device__ __forceinline__ v16h load_b_f32(const float* col_base, int ld) {
    int l = threadIdx.x & 31; int n = l & 15; int hs = l >> 4;
    const float* p = col_base + (size_t)n * ld + hs * 16;
    v16h b;
#pragma unroll
    for (int i = 0; i < 16; ++i) b[i] = (_Float16)p[i];
    return b;
}
__device__ __forceinline__ v16h load_b_f16(const _Float16* col_base, int ld) {
    int l = threadIdx.x & 31; int n = l & 15; int hs = l >> 4;
    const _Float16* p = col_base + (size_t)n * ld + hs * 16;
    v16h b;
#pragma unroll
    for (int i = 0; i < 16; ++i) b[i] = p[i];
    return b;
}

// =====================================================================
// Kernel 1: QKV projections.  out = X @ W^T + bias  (f32 in, f16 out)
//   Register-blocked: each wave computes a 16(M) x 64(N) strip — one A
//   tile reused against 4 B tiles (4 WMMAs per A load).
//   blockIdx.z: 0 -> Q (from x), 1 -> K (from attend_over), 2 -> V (transposed)
//   grid (M/64, N/64, 3), block 128 (4 waves: wave w owns M-subtile w)
// =====================================================================
__global__ void __launch_bounds__(128)
proj_kernel(const float* __restrict__ x, const float* __restrict__ ao,
            const float* __restrict__ Wq, const float* __restrict__ bq,
            const float* __restrict__ Wk, const float* __restrict__ bk,
            const float* __restrict__ Wv, const float* __restrict__ bv,
            _Float16* __restrict__ qh, _Float16* __restrict__ kh,
            _Float16* __restrict__ vt) {
    int which = blockIdx.z;
    const float* X; const float* W; const float* bias;
    if (which == 0)      { X = x;  W = Wq; bias = bq; }
    else if (which == 1) { X = ao; W = Wk; bias = bk; }
    else                 { X = ao; W = Wv; bias = bv; }

    int wave = threadIdx.x >> 5, lane = threadIdx.x & 31;
    int m0 = blockIdx.x * 64 + wave * 16;
    int n0 = blockIdx.y * 64;

    v8f acc[4] = {};
#pragma unroll 2
    for (int kk = 0; kk < D; kk += 32) {
        v16h a = load_a_f32(X + (size_t)m0 * D + kk, D);
#pragma unroll
        for (int t = 0; t < 4; ++t) {
            v16h b = load_b_f32(W + (size_t)(n0 + 16 * t) * D + kk, D); // B[k][n]=W[n][k]
            acc[t] = wmma_f16(a, b, acc[t]);
        }
    }

    int nl = lane & 15, hf = lane >> 4;
#pragma unroll
    for (int t = 0; t < 4; ++t) {
        int n = n0 + 16 * t + nl;
        float bn = bias[n];
        int hh = n >> 6, d = n & (DK - 1);
#pragma unroll
        for (int r = 0; r < 8; ++r) {
            int mrow = m0 + r + 8 * hf;
            int bb = mrow >> 11;           // / S
            int s  = mrow & (S - 1);       // % S
            float v = acc[t][r] + bn;
            if (which == 0)
                qh[(((size_t)bb * H + hh) * S + s) * DK + d] = (_Float16)v;
            else if (which == 1)
                kh[(((size_t)bb * H + hh) * S + s) * DK + d] = (_Float16)v;
            else
                vt[(((size_t)bb * H + hh) * DK + d) * S + s] = (_Float16)v; // transposed
        }
    }
}

// =====================================================================
// Kernel 2: attention. One block per (b, h, 16-query tile).
//   256 threads = 8 waves. Full 16 x 2048 f32 score panel lives in LDS
//   (128 KB — CDNA5 supports up to 320 KB/workgroup).
// =====================================================================
__global__ void __launch_bounds__(256)
attn_kernel(const _Float16* __restrict__ qh, const _Float16* __restrict__ kh,
            const _Float16* __restrict__ vt, const int* __restrict__ amask,
            float* __restrict__ attn_out, _Float16* __restrict__ ctx_h) {
    extern __shared__ float smem[];
    float* sc   = smem;                 // [16][S]   scores -> exp -> probs
    float* red  = sc + 16 * S;          // [256]     reduction scratch
    float* rowv = red + 256;            // [16]      row max, then 1/sum
    float* ctxb = rowv + 16;            // [16][64]  ctx accumulation

    int blk = blockIdx.x;
    int qt = blk & 127; int h = (blk >> 7) & 7; int b = blk >> 10;
    int tid = threadIdx.x, wave = tid >> 5, lane = tid & 31;
    int hf = lane >> 4;

    // -------- per-wave copy of the Q tile (16 x 64, two K=32 chunks) --------
    const _Float16* qbase = qh + (((size_t)b * H + h) * S + qt * 16) * DK;
    v16h qa0 = load_a_f16(qbase,      DK);
    v16h qa1 = load_a_f16(qbase + 32, DK);

    // -------- scores: each wave strides over key tiles --------
    const _Float16* kbase = kh + ((size_t)b * H + h) * S * DK;
    for (int kt = wave; kt < S / 16; kt += 8) {
        const _Float16* kb = kbase + (size_t)kt * 16 * DK;
        v16h b0 = load_b_f16(kb,      DK);     // B[d][n] = k[n][d], d in [0,32)
        v16h b1 = load_b_f16(kb + 32, DK);     // d in [32,64)
        v8f acc = {};
        acc = wmma_f16(qa0, b0, acc);
        acc = wmma_f16(qa1, b1, acc);

        int kcol = kt * 16 + (lane & 15);
        int mv = amask[b * S + kcol];
#pragma unroll
        for (int r = 0; r < 8; ++r) {
            int qrow = qt * 16 + r + 8 * hf;
            float v = (mv != 0 && kcol <= qrow) ? acc[r] * INV_TEMP : NEG;
            sc[(size_t)(r + 8 * hf) * S + kcol] = v;
        }
    }
    __syncthreads();

    // -------- fp32 softmax: row g handled by 16-thread group --------
    int g = tid >> 4, li = tid & 15;
    float mx = -3.4e38f;
    for (int k = li; k < S; k += 16) mx = fmaxf(mx, sc[(size_t)g * S + k]);
    red[tid] = mx;
    __syncthreads();
    if (li == 0) {
        float t = red[g * 16];
#pragma unroll
        for (int i = 1; i < 16; ++i) t = fmaxf(t, red[g * 16 + i]);
        rowv[g] = t;
    }
    __syncthreads();
    float rmax = rowv[g];
    float sum = 0.f;
    for (int k = li; k < S; k += 16) {
        float e = __expf(sc[(size_t)g * S + k] - rmax);
        sc[(size_t)g * S + k] = e;
        sum += e;
    }
    red[tid] = sum;
    __syncthreads();
    if (li == 0) {
        float t = 0.f;
#pragma unroll
        for (int i = 0; i < 16; ++i) t += red[g * 16 + i];
        rowv[g] = 1.0f / t;
    }
    __syncthreads();

    // -------- normalize, write attn (b, q, h, k) coalesced --------
    float* aout = attn_out + (((size_t)b * S + qt * 16) * H + h) * S;
    for (int idx = tid; idx < 16 * S; idx += 256) {
        int row = idx >> 11;           // / S
        int k   = idx & (S - 1);       // % S
        float p = sc[(size_t)row * S + k] * rowv[row];
        sc[(size_t)row * S + k] = p;
        aout[(size_t)row * H * S + k] = p;
    }
    __syncthreads();

    // -------- ctx = probs @ V : split-K over wave pairs --------
    int dt = wave & 3, khalf = wave >> 2;
    const _Float16* vb = vt + (((size_t)b * H + h) * DK + dt * 16) * S;
    v8f cacc = {};
    for (int kt = khalf * 32; kt < khalf * 32 + 32; ++kt) {
        v16h pa = load_a_f32(sc + kt * 32, S);           // probs f32 -> f16
        v16h vv = load_b_f16(vb + kt * 32, S);           // B[k][d] = vt[d][k]
        cacc = wmma_f16(pa, vv, cacc);
    }
    if (khalf == 0) {
#pragma unroll
        for (int r = 0; r < 8; ++r)
            ctxb[(r + 8 * hf) * 64 + dt * 16 + (lane & 15)] = cacc[r];
    }
    __syncthreads();
    if (khalf == 1) {
#pragma unroll
        for (int r = 0; r < 8; ++r)
            ctxb[(r + 8 * hf) * 64 + dt * 16 + (lane & 15)] += cacc[r];
    }
    __syncthreads();

    // -------- emit ctx tile as f16 (B, S, D) --------
    for (int idx = tid; idx < 16 * 64; idx += 256) {
        int row = idx >> 6, d = idx & 63;
        int s = qt * 16 + row;
        ctx_h[((size_t)b * S + s) * D + h * DK + d] = (_Float16)ctxb[idx];
    }
}

// =====================================================================
// Kernel 3: output projection. context = ctx @ Wo^T + bo  (f32 out)
//   Same 16x64-per-wave register blocking as proj_kernel.
// =====================================================================
__global__ void __launch_bounds__(128)
oproj_kernel(const _Float16* __restrict__ ctx_h, const float* __restrict__ Wo,
             const float* __restrict__ bo, float* __restrict__ out) {
    int wave = threadIdx.x >> 5, lane = threadIdx.x & 31;
    int m0 = blockIdx.x * 64 + wave * 16;
    int n0 = blockIdx.y * 64;

    v8f acc[4] = {};
#pragma unroll 2
    for (int kk = 0; kk < D; kk += 32) {
        v16h a = load_a_f16(ctx_h + (size_t)m0 * D + kk, D);
#pragma unroll
        for (int t = 0; t < 4; ++t) {
            v16h b = load_b_f32(Wo + (size_t)(n0 + 16 * t) * D + kk, D);
            acc[t] = wmma_f16(a, b, acc[t]);
        }
    }
    int nl = lane & 15, hf = lane >> 4;
#pragma unroll
    for (int t = 0; t < 4; ++t) {
        int n = n0 + 16 * t + nl;
        float bn = bo[n];
#pragma unroll
        for (int r = 0; r < 8; ++r) {
            int mrow = m0 + r + 8 * hf;
            out[(size_t)mrow * D + n] = acc[t][r] + bn;
        }
    }
}

// =====================================================================
// Host launcher
// =====================================================================
extern "C" void kernel_launch(void* const* d_in, const int* in_sizes, int n_in,
                              void* d_out, int out_size, void* d_ws, size_t ws_size,
                              hipStream_t stream) {
    // setup_inputs order:
    // 0:x 1:attend_over 2:attend_mask 3:Wk 4:bk 5:Wv 6:bv 7:Wq 8:bq 9:Wo 10:bo
    const float* x    = (const float*)d_in[0];
    const float* ao   = (const float*)d_in[1];
    const int*   msk  = (const int*)  d_in[2];
    const float* Wk   = (const float*)d_in[3];
    const float* bk   = (const float*)d_in[4];
    const float* Wv   = (const float*)d_in[5];
    const float* bv   = (const float*)d_in[6];
    const float* Wq   = (const float*)d_in[7];
    const float* bq   = (const float*)d_in[8];
    const float* Wo   = (const float*)d_in[9];
    const float* bo   = (const float*)d_in[10];

    float* out_ctx  = (float*)d_out;                       // (B,S,D)
    float* out_attn = (float*)d_out + (size_t)B * S * D;   // (B,S,H,S)

    // workspace: 4 f16 tensors of B*S*D elements (4 MB each, 16 MB total)
    _Float16* qh    = (_Float16*)d_ws;
    _Float16* kh    = qh + (size_t)B * S * D;
    _Float16* vt    = kh + (size_t)B * S * D;
    _Float16* ctxh  = vt + (size_t)B * S * D;

    // 1) projections: M=B*S=4096 -> 64 blocks of 64, N=512 -> 8 blocks of 64, z = Q/K/V
    dim3 gp(64, 8, 3);
    proj_kernel<<<gp, 128, 0, stream>>>(x, ao, Wq, bq, Wk, bk, Wv, bv, qh, kh, vt);

    // 2) attention: one block per (b,h,q-tile); 133 KB dynamic LDS (<< 320 KB/WGP)
    size_t lds_bytes = (size_t)(16 * S + 256 + 16 + 16 * 64) * sizeof(float);
    attn_kernel<<<B * H * (S / 16), 256, lds_bytes, stream>>>(
        qh, kh, vt, msk, out_attn, ctxh);

    // 3) output projection -> f32 context
    dim3 go(64, 8);
    oproj_kernel<<<go, 128, 0, stream>>>(ctxh, Wo, bo, out_ctx);
}
